// BitLinear_70677981823273
// MI455X (gfx1250) — matrix operand order repaired and checked by reading, not compiled, not run on hardware
//
#include <hip/hip_runtime.h>
#include <stdint.h>

// ---------------------------------------------------------------------------
// BitLinear (BitNet 1.58) forward for MI455X / gfx1250 (wave32, WMMA).
//   out[m,n] = sum_k x[m,k] * Wq[n,k] + bias[n],  Wq = clamp(round(W/gamma),-1,1)
// M=16384 (=4*4096), N=4096, K=4096, all fp32 in/out.
//
// Roofline: 0.55 TFLOP GEMM, ~0.58 GB min HBM traffic (~25 us @ 23.3 TB/s)
// vs ~110 us at bf16-WMMA dense peak -> compute bound. Ternary weights are
// exact in bf16; only x->bf16 rounding deviates from the fp32 reference.
//
// Fast path (if ws_size allows): precompute x (bf16) and Wq (bf16) into the
// workspace; the GEMM stages tiles with GLOBAL_LOAD_ASYNC_TO_LDS_B128
// (ASYNCcnt-tracked, no VGPR round trip) when the toolchain exposes it.
// Fallback: convert/ternarize on the fly while staging through registers.
// Both GEMMs use ping-pong LDS (1 barrier/iter, staging overlaps WMMA).
// ---------------------------------------------------------------------------

typedef __attribute__((ext_vector_type(16))) __bf16 v16bf;
typedef __attribute__((ext_vector_type(8)))  float  v8f;
typedef __attribute__((ext_vector_type(4)))  int    v4i;

#define M_DIM 16384
#define N_DIM 4096
#define K_DIM 4096

// --- async global->LDS staging availability (gfx1250; clang-22+) -----------
#if defined(__has_builtin)
#if __has_builtin(__builtin_amdgcn_global_load_async_to_lds_b128) && \
    __has_builtin(__builtin_amdgcn_s_wait_asynccnt)
#define ASYNC_LDS 1
#endif
#endif
#ifndef ASYNC_LDS
#define ASYNC_LDS 0
#endif

#if ASYNC_LDS
__device__ __forceinline__ void async_copy_b128(const void* g, void* l) {
    // Builtin signature (per hipcc diagnostic): int4-vector pointers.
    __builtin_amdgcn_global_load_async_to_lds_b128(
        (v4i*)(void*)g, (v4i*)l, /*offset=*/0, /*cpol=*/0);
}
#endif

// fp32 pair -> packed bf16 (RNE). Prefer the gfx1250 v_cvt_pk_bf16_f32 op.
#if defined(__has_builtin) && __has_builtin(__builtin_amdgcn_cvt_pk_bf16_f32)
typedef __attribute__((ext_vector_type(2))) __bf16 v2bf;
__device__ __forceinline__ unsigned pack_bf16(float a, float b) {
    v2bf r = __builtin_amdgcn_cvt_pk_bf16_f32(a, b);
    unsigned u;
    __builtin_memcpy(&u, &r, 4);
    return u;
}
#else
__device__ __forceinline__ unsigned pack_bf16(float a, float b) {
    unsigned ua = __float_as_uint(a);
    ua = ua + 0x7FFFu + ((ua >> 16) & 1u);
    unsigned ub = __float_as_uint(b);
    ub = ub + 0x7FFFu + ((ub >> 16) & 1u);
    return (ua >> 16) | (ub & 0xFFFF0000u);
}
#endif

// BitNet absmean ternarization (fminf/fmaxf folds to v_med3_f32).
__device__ __forceinline__ float ternarize(float wv, float inv_gamma) {
    float q = rintf(wv * inv_gamma);
    return fminf(1.0f, fmaxf(-1.0f, q));
}

union FragU {
    uint4 u[2];
    v16bf v;
};

// ---------------------------------------------------------------------------
// Kernel 1: zero the gamma accumulator (ws is poisoned by the harness).
// ---------------------------------------------------------------------------
__global__ void init_ws_kernel(float* __restrict__ gsum) {
    if (threadIdx.x == 0 && blockIdx.x == 0) gsum[0] = 0.0f;
}

// ---------------------------------------------------------------------------
// Kernel 2: sum(|W|) over 4096x4096 via grid-stride + LDS reduce + atomicAdd.
// ---------------------------------------------------------------------------
__global__ __launch_bounds__(256)
void absmean_kernel(const float* __restrict__ w, float* __restrict__ gsum) {
    __shared__ float red[256];
    const float4* w4 = (const float4*)w;
    const size_t n4 = (size_t)N_DIM * K_DIM / 4;
    size_t i = (size_t)blockIdx.x * blockDim.x + threadIdx.x;
    const size_t stride = (size_t)gridDim.x * blockDim.x;
    float s = 0.0f;
    for (; i < n4; i += stride) {
        float4 v = w4[i];
        s += fabsf(v.x) + fabsf(v.y) + fabsf(v.z) + fabsf(v.w);
    }
    red[threadIdx.x] = s;
    __syncthreads();
    for (int o = 128; o > 0; o >>= 1) {
        if ((int)threadIdx.x < o) red[threadIdx.x] += red[threadIdx.x + o];
        __syncthreads();
    }
    if (threadIdx.x == 0) atomicAdd(gsum, red[0]);
}

// ---------------------------------------------------------------------------
// Kernel 3a: x fp32 -> bf16 (packed pairs) into workspace.
// ---------------------------------------------------------------------------
__global__ __launch_bounds__(256)
void convert_x_kernel(const float* __restrict__ x, unsigned* __restrict__ xq) {
    const float4* x4 = (const float4*)x;
    uint2* q2 = (uint2*)xq;
    const size_t n4 = (size_t)M_DIM * K_DIM / 4;
    size_t i = (size_t)blockIdx.x * blockDim.x + threadIdx.x;
    const size_t stride = (size_t)gridDim.x * blockDim.x;
    for (; i < n4; i += stride) {
        float4 v = x4[i];
        q2[i] = make_uint2(pack_bf16(v.x, v.y), pack_bf16(v.z, v.w));
    }
}

// ---------------------------------------------------------------------------
// Kernel 3b: W fp32 -> ternary bf16 (exact) into workspace.
// ---------------------------------------------------------------------------
__global__ __launch_bounds__(256)
void quantize_w_kernel(const float* __restrict__ w, unsigned* __restrict__ wq,
                       const float* __restrict__ gsum) {
    const float gamma = gsum[0] * (1.0f / (float)((size_t)N_DIM * K_DIM)) + 1e-8f;
    const float inv_gamma = 1.0f / gamma;
    const float4* w4 = (const float4*)w;
    uint2* q2 = (uint2*)wq;
    const size_t n4 = (size_t)N_DIM * K_DIM / 4;
    size_t i = (size_t)blockIdx.x * blockDim.x + threadIdx.x;
    const size_t stride = (size_t)gridDim.x * blockDim.x;
    for (; i < n4; i += stride) {
        float4 v = w4[i];
        q2[i] = make_uint2(
            pack_bf16(ternarize(v.x, inv_gamma), ternarize(v.y, inv_gamma)),
            pack_bf16(ternarize(v.z, inv_gamma), ternarize(v.w, inv_gamma)));
    }
}

// ---------------------------------------------------------------------------
// Kernel 4: tiled GEMM with v_wmma_f32_16x16x32_bf16, ping-pong LDS.
//   Block tile 128x128, 256 threads = 8 waves (4 along M x 2 along N).
//   Each wave: 2x4 accumulator tiles of 16x16 (v8f). BK = 32.
//   PRE=true : bf16 tiles from workspace; async global->LDS if available.
//   PRE=false: fp32 inputs, convert/ternarize while staging via registers.
// LDS rows are bf16 pairs as uint32, row stride 20 uints (80 B) -- all
// fragment reads are 16B-aligned ds_load_b128 per ISA 7.12.2 layouts.
// ---------------------------------------------------------------------------
template <bool PRE>
__global__ __launch_bounds__(256)
void bitlinear_gemm_kernel(const float* __restrict__ x,
                           const float* __restrict__ w,
                           const unsigned* __restrict__ xq,
                           const unsigned* __restrict__ wq,
                           const float* __restrict__ bias,
                           const float* __restrict__ gsum,
                           float* __restrict__ out) {
    __shared__ unsigned lds_x[2][128 * 20];   // 2 x 10 KB ping-pong
    __shared__ unsigned lds_w[2][128 * 20];   // 2 x 10 KB ping-pong

    const int tid   = threadIdx.x;
    const int lane  = tid & 31;
    const int wave  = tid >> 5;
    const int waveM = wave & 3;             // 0..3 (M)
    const int waveN = wave >> 2;            // 0..1 (N)
    const int lr    = lane & 15;
    const int hs    = (lane >> 4) & 1;

    const int n0 = blockIdx.x * 128;
    const int m0 = blockIdx.y * 128;

    float inv_gamma = 0.0f;
    if constexpr (!PRE) {
        const float gamma =
            gsum[0] * (1.0f / (float)((size_t)N_DIM * K_DIM)) + 1e-8f;
        inv_gamma = 1.0f / gamma;
    }

    // Cooperative staging: thread t handles row t/2, 16-element segment t%2.
    const int ldrow = tid >> 1;
    const int ldseg = tid & 1;
    const size_t xrow = (size_t)(m0 + ldrow) * K_DIM;
    const size_t wrow = (size_t)(n0 + ldrow) * K_DIM;
    const float*    xsrc  = x + xrow + ldseg * 16;
    const float*    wsrc  = w + wrow + ldseg * 16;
    const unsigned* xqsrc = xq ? xq + (xrow >> 1) + ldseg * 8 : nullptr;
    const unsigned* wqsrc = wq ? wq + (wrow >> 1) + ldseg * 8 : nullptr;
    const int ldst = ldrow * 20 + ldseg * 8;

    v8f acc[2][4];
    const v8f vzero = {0.f, 0.f, 0.f, 0.f, 0.f, 0.f, 0.f, 0.f};
    for (int i = 0; i < 2; ++i)
        for (int j = 0; j < 4; ++j) acc[i][j] = vzero;

    // Fragment base offsets (uints) per ISA 7.12.2:
    // A (16x32 bf16): lanes 0-15 K0-7 & K16-23; lanes 16-31 K8-15 & K24-31.
    // B (32x16 bf16): lanes 0-15 K0-7 & K16-23; lanes 16-31 K8-15 & K24-31.
    int aoff[2], boff[4];
#pragma unroll
    for (int tm = 0; tm < 2; ++tm)
        aoff[tm] = (waveM * 32 + tm * 16 + lr) * 20 + hs * 4;
#pragma unroll
    for (int tn = 0; tn < 4; ++tn)
        boff[tn] = (waveN * 64 + tn * 16 + lr) * 20 + hs * 4;

    // Fragment build + 8 WMMA for one staged tile.
    auto compute_tile = [&](const unsigned* lx, const unsigned* lw) {
        v16bf afr[2], bfr[4];
#pragma unroll
        for (int tm = 0; tm < 2; ++tm) {
            FragU f;
            f.u[0] = *(const uint4*)&lx[aoff[tm] + 0];
            f.u[1] = *(const uint4*)&lx[aoff[tm] + 8];
            afr[tm] = f.v;
        }
#pragma unroll
        for (int tn = 0; tn < 4; ++tn) {
            FragU f;
            f.u[0] = *(const uint4*)&lw[boff[tn] + 0];
            f.u[1] = *(const uint4*)&lw[boff[tn] + 8];
            bfr[tn] = f.v;
        }
#pragma unroll
        for (int tm = 0; tm < 2; ++tm)
#pragma unroll
            for (int tn = 0; tn < 4; ++tn)
                acc[tm][tn] = __builtin_amdgcn_wmma_f32_16x16x32_bf16(
                    false, afr[tm], false, bfr[tn],
                    (short)0, acc[tm][tn], false, false);
    };

    const int NK = K_DIM / 32;

#if ASYNC_LDS
    if constexpr (PRE) {
        // ---- async global->LDS pipeline (ASYNCcnt-tracked DMA copies) ----
        auto async_stage = [&](int it, int b) {
            const unsigned* gx = xqsrc + (it * 16);   // 32 K-elems = 16 uints
            const unsigned* gw = wqsrc + (it * 16);
            async_copy_b128(gx + 0, &lds_x[b][ldst + 0]);
            async_copy_b128(gx + 4, &lds_x[b][ldst + 4]);
            async_copy_b128(gw + 0, &lds_w[b][ldst + 0]);
            async_copy_b128(gw + 4, &lds_w[b][ldst + 4]);
        };

        async_stage(0, 0);
        int buf = 0;
        for (int it = 0; it < NK; ++it) {
            __builtin_amdgcn_s_wait_asynccnt(0);  // my tile-`it` bytes landed
            __syncthreads();                      // everyone's landed; buf^1 free
            if (it + 1 < NK) async_stage(it + 1, buf ^ 1);
            compute_tile(lds_x[buf], lds_w[buf]);
            buf ^= 1;
        }
    } else
#endif
    {
        // ---- register-staging pipeline (with on-the-fly conversion) ----
        auto stage_x = [&](int k0, uint4& lo, uint4& hi) {
            if constexpr (PRE) {
                lo = *(const uint4*)(xqsrc + (k0 >> 1));
                hi = *(const uint4*)(xqsrc + (k0 >> 1) + 4);
            } else {
                float4 a = *(const float4*)(xsrc + k0 + 0);
                float4 b = *(const float4*)(xsrc + k0 + 4);
                float4 c = *(const float4*)(xsrc + k0 + 8);
                float4 d = *(const float4*)(xsrc + k0 + 12);
                lo = make_uint4(pack_bf16(a.x, a.y), pack_bf16(a.z, a.w),
                                pack_bf16(b.x, b.y), pack_bf16(b.z, b.w));
                hi = make_uint4(pack_bf16(c.x, c.y), pack_bf16(c.z, c.w),
                                pack_bf16(d.x, d.y), pack_bf16(d.z, d.w));
            }
        };
        auto stage_w = [&](int k0, uint4& lo, uint4& hi) {
            if constexpr (PRE) {
                lo = *(const uint4*)(wqsrc + (k0 >> 1));
                hi = *(const uint4*)(wqsrc + (k0 >> 1) + 4);
            } else {
                float4 a = *(const float4*)(wsrc + k0 + 0);
                float4 b = *(const float4*)(wsrc + k0 + 4);
                float4 c = *(const float4*)(wsrc + k0 + 8);
                float4 d = *(const float4*)(wsrc + k0 + 12);
                lo = make_uint4(
                    pack_bf16(ternarize(a.x, inv_gamma), ternarize(a.y, inv_gamma)),
                    pack_bf16(ternarize(a.z, inv_gamma), ternarize(a.w, inv_gamma)),
                    pack_bf16(ternarize(b.x, inv_gamma), ternarize(b.y, inv_gamma)),
                    pack_bf16(ternarize(b.z, inv_gamma), ternarize(b.w, inv_gamma)));
                hi = make_uint4(
                    pack_bf16(ternarize(c.x, inv_gamma), ternarize(c.y, inv_gamma)),
                    pack_bf16(ternarize(c.z, inv_gamma), ternarize(c.w, inv_gamma)),
                    pack_bf16(ternarize(d.x, inv_gamma), ternarize(d.y, inv_gamma)),
                    pack_bf16(ternarize(d.z, inv_gamma), ternarize(d.w, inv_gamma)));
            }
        };

        {   // prologue: stage tile 0 into buffer 0
            uint4 xlo, xhi, wlo, whi;
            stage_x(0, xlo, xhi);
            stage_w(0, wlo, whi);
            *(uint4*)&lds_x[0][ldst + 0] = xlo;
            *(uint4*)&lds_x[0][ldst + 4] = xhi;
            *(uint4*)&lds_w[0][ldst + 0] = wlo;
            *(uint4*)&lds_w[0][ldst + 4] = whi;
        }

        int buf = 0;
        for (int it = 0; it < NK; ++it) {
            const bool has_next = (it + 1) < NK;
            uint4 nxlo, nxhi, nwlo, nwhi;
            if (has_next) {     // prefetch tile it+1 (overlaps WMMA below)
                stage_x((it + 1) * 32, nxlo, nxhi);
                stage_w((it + 1) * 32, nwlo, nwhi);
            }
            __syncthreads();    // tile `it` visible; buf^1 readers done
            compute_tile(lds_x[buf], lds_w[buf]);
            if (has_next) {
                unsigned* lxn = lds_x[buf ^ 1];
                unsigned* lwn = lds_w[buf ^ 1];
                *(uint4*)&lxn[ldst + 0] = nxlo;
                *(uint4*)&lxn[ldst + 4] = nxhi;
                *(uint4*)&lwn[ldst + 0] = nwlo;
                *(uint4*)&lwn[ldst + 4] = nwhi;
            }
            buf ^= 1;
        }
    }

    // ---- epilogue: C/D layout (VGPR r -> row r / r+8, lane -> column) ----
#pragma unroll
    for (int tm = 0; tm < 2; ++tm) {
#pragma unroll
        for (int tn = 0; tn < 4; ++tn) {
            const int col = n0 + waveN * 64 + tn * 16 + lr;
            const float bv = bias[col];
            const int r0 = m0 + waveM * 32 + tm * 16 + hs * 8;
            union { v8f v; float f[8]; } cu;
            cu.v = acc[tm][tn];
#pragma unroll
            for (int r = 0; r < 8; ++r)
                out[(size_t)(r0 + r) * N_DIM + col] = cu.f[r] + bv;
        }
    }
}

// ---------------------------------------------------------------------------
extern "C" void kernel_launch(void* const* d_in, const int* in_sizes, int n_in,
                              void* d_out, int out_size, void* d_ws, size_t ws_size,
                              hipStream_t stream) {
    const float* x    = (const float*)d_in[0];   // [4,4096,4096] fp32
    const float* w    = (const float*)d_in[1];   // [4096,4096]   fp32
    const float* bias = (const float*)d_in[2];   // [4096]        fp32
    float* out  = (float*)d_out;                 // [4,4096,4096] fp32
    float* gsum = (float*)d_ws;

    const size_t x_bytes = (size_t)M_DIM * K_DIM * 2;   // 128 MB bf16
    const size_t w_bytes = (size_t)N_DIM * K_DIM * 2;   //  32 MB bf16
    const size_t ws_need = 256 + x_bytes + w_bytes;

    init_ws_kernel<<<1, 1, 0, stream>>>(gsum);
    absmean_kernel<<<1024, 256, 0, stream>>>(w, gsum);

    dim3 grid(N_DIM / 128, M_DIM / 128);         // (32, 128)

    if (ws_size >= ws_need) {
        // Fast path: hoist all conversions out of the GEMM inner loop.
        unsigned* xq = (unsigned*)((char*)d_ws + 256);
        unsigned* wq = (unsigned*)((char*)d_ws + 256 + x_bytes);
        convert_x_kernel<<<4096, 256, 0, stream>>>(x, xq);
        quantize_w_kernel<<<1024, 256, 0, stream>>>(w, wq, gsum);
        bitlinear_gemm_kernel<true><<<grid, 256, 0, stream>>>(
            x, w, xq, wq, bias, gsum, out);
    } else {
        // Fallback: convert/ternarize while staging to LDS (4 bytes of ws).
        bitlinear_gemm_kernel<false><<<grid, 256, 0, stream>>>(
            x, w, nullptr, nullptr, bias, gsum, out);
    }
}
